// Head_21758304321795
// MI455X (gfx1250) — matrix operand order repaired and checked by reading, not compiled, not run on hardware
//
#include <hip/hip_runtime.h>
#include <hip/hip_bf16.h>
#include <math.h>

// ---------------------------------------------------------------------------
// One head of "additive tanh" attention, collapsed algebraically:
//   q_w + k_w = x @ (W_q@W_ql + W_k@W_kl) = x @ M     (elementwise add trick)
//   out_b = softmax(causal(tanh(x_b @ M))) @ (x_b @ W_v)
// Memory-bound target: read x (201 MB) once, write out (33.5 MB) once.
// All GEMMs via v_wmma_f32_16x16x32_bf16 (wave32, 16x16 tiles, K=32).
// Round-2 changes: single per-lane base pointer per stream + fully unrolled
// K loops so n/kk offsets fold into the 24-bit global ioffset (kills the
// v_add_nc_u64 chains); softmax drops the max pass (tanh-bounded scores) and
// uses hardware v_tanh_f32 when available.
// ---------------------------------------------------------------------------

typedef __bf16 bf16;
typedef __attribute__((ext_vector_type(16))) __bf16 v16bf;
typedef __attribute__((ext_vector_type(8)))  __bf16 v8bf;
typedef __attribute__((ext_vector_type(8)))  float  v8f;

#if __has_builtin(__builtin_amdgcn_tanhf)
#define TANHF(x) __builtin_amdgcn_tanhf(x)
#elif __has_builtin(__builtin_amdgcn_tanh_f32)
#define TANHF(x) __builtin_amdgcn_tanh_f32(x)
#else
#define TANHF(x) tanhf(x)
#endif

#define Bn   512
#define Tn   256
#define Cn   384
#define HSn  64
#define KS_C 12          // 384 / 32 k-steps for the C-reduction GEMMs
#define KS_T 8           // 256 / 32 k-steps for P@V
#define P_STRIDE  264    // 256 + 8 bf16 -> 132 dwords == 4 mod 64 (bank-safe)
#define VT_STRIDE 264
#define SMEM_ELTS (Tn * P_STRIDE + HSn * VT_STRIDE)   // 165 KB of LDS

// -------------------- prep: Mt[n][k] = (Wq@Wql + Wk@Wkl)^T, WvT[h][c] -------
__global__ void prep_kernel(const float* __restrict__ Wq, const float* __restrict__ Wk,
                            const float* __restrict__ Wv, const float* __restrict__ Wql,
                            const float* __restrict__ Wkl,
                            bf16* __restrict__ Mt, bf16* __restrict__ WvT) {
    int idx = blockIdx.x * blockDim.x + threadIdx.x;
    if (idx < Tn * Cn) {                 // Mt[n][k], n in [0,256), k in [0,384)
        int n = idx / Cn;
        int k = idx % Cn;
        float acc = 0.f;
        #pragma unroll 8
        for (int h = 0; h < HSn; ++h)
            acc += Wq[k * HSn + h] * Wql[h * Tn + n]
                 + Wk[k * HSn + h] * Wkl[h * Tn + n];
        Mt[idx] = (bf16)acc;
    } else {
        int j = idx - Tn * Cn;
        if (j < HSn * Cn) {              // WvT[h][c] = Wv[c][h]
            int h = j / Cn, c = j % Cn;
            WvT[j] = (bf16)Wv[c * HSn + h];
        }
    }
}

// Concatenate two 8-element bf16 chunks into one WMMA operand register group.
__device__ __forceinline__ v16bf cat8(v8bf lo, v8bf hi) {
    return __builtin_shufflevector(lo, hi, 0, 1, 2, 3, 4, 5, 6, 7,
                                           8, 9, 10, 11, 12, 13, 14, 15);
}

// -------------------- main: one batch element per workgroup ------------------
__global__ __launch_bounds__(512, 1) void head_main(
    const float* __restrict__ x, const bf16* __restrict__ Mt,
    const bf16* __restrict__ WvT, float* __restrict__ out) {
    extern __shared__ bf16 smem[];
    bf16* sP  = smem;                    // probabilities, [256][P_STRIDE]
    bf16* sVT = smem + Tn * P_STRIDE;    // V^T,           [64][VT_STRIDE]

    const int b    = blockIdx.x;
    const int tid  = threadIdx.x;
    const int wid  = tid >> 5;           // 16 waves
    const int lane = tid & 31;
    const int half = lane >> 4;          // C-layout: rows 0-7 in lanes 0-15
    const int nlo  = lane & 15;
    const int t0   = wid * 16;           // this wave's 16-row strip

    // A-operand K-chunk split (contiguous, matches B):
    //   lanes<16 -> K 0..15, lanes>=16 -> K 16..31.
    const int aoff = half ? 16 : 0;
    // B-operand (column-major source): lanes<16 -> K 0..15, lanes>=16 -> K 16..31.
    const int boff = half ? 16 : 0;

    // ONE per-lane base pointer per stream; all n/kk variation becomes a
    // compile-time byte offset folded into the load instruction's ioffset.
    const float* xp  = x + ((size_t)b * Tn + t0 + nlo) * Cn + aoff;
    const bf16*  mtp = Mt  + (size_t)nlo * Cn + boff;    // + n*16*Cn + kk*32 (imm)
    const bf16*  wvp = WvT + (size_t)nlo * Cn + boff;

    v8f accS[16];                        // 16x256 score strip, fp32
    v8f accV[4];                         // 16x64 V strip
    #pragma unroll
    for (int n = 0; n < 16; ++n) accS[n] = (v8f)0.f;
    #pragma unroll
    for (int n = 0; n < 4; ++n) accV[n] = (v8f)0.f;

    // ---- GEMM 1+2 fused over K=C: S = x@M (16 tiles), V = x@Wv (4 tiles) ----
    #pragma unroll
    for (int kk = 0; kk < KS_C; ++kk) {
        v8f f0 = *(const v8f*)(xp + kk * 32);
        v8f f1 = *(const v8f*)(xp + kk * 32 + 8);
        v16bf a = cat8(__builtin_convertvector(f0, v8bf),
                       __builtin_convertvector(f1, v8bf));
        #pragma unroll
        for (int n = 0; n < 16; ++n) {
            const bf16* colb = mtp + n * 16 * Cn + kk * 32;   // imm: n*12288+kk*64 B
            v16bf bt = cat8(*(const v8bf*)colb, *(const v8bf*)(colb + 8));
            accS[n] = __builtin_amdgcn_wmma_f32_16x16x32_bf16(
                false, a, false, bt, (short)0, accS[n], false, false);
        }
        #pragma unroll
        for (int n = 0; n < 4; ++n) {
            const bf16* colb = wvp + n * 16 * Cn + kk * 32;
            v16bf bt = cat8(*(const v8bf*)colb, *(const v8bf*)(colb + 8));
            accV[n] = __builtin_amdgcn_wmma_f32_16x16x32_bf16(
                false, a, false, bt, (short)0, accV[n], false, false);
        }
    }

    // ---- scatter V^T (bf16) to LDS: element (s = t0+j+8*half, h = n*16+nlo)
    {
        bf16* vbase = sVT + (size_t)nlo * VT_STRIDE + (t0 + 8 * half);
        #pragma unroll
        for (int n = 0; n < 4; ++n)
            #pragma unroll
            for (int j = 0; j < 8; ++j)
                vbase[n * 16 * VT_STRIDE + j] = (bf16)accV[n][j];
    }

    // ---- tanh + causal mask + softmax ---------------------------------------
    // Scores are tanh-bounded in [-1,1], so exp cannot overflow: skip the
    // max-subtraction pass entirely.  Row t = t0 + j + 8*half lives in VGPR j
    // across lanes; xor-shuffles (8,4,2,1) reduce within each 16-lane half.
    #pragma unroll
    for (int j = 0; j < 8; ++j) {
        const int t = t0 + j + 8 * half;
        float sum = 0.f;
        #pragma unroll
        for (int n = 0; n < 16; ++n) {
            int s = n * 16 + nlo;
            float e = (s <= t) ? __expf(TANHF(accS[n][j])) : 0.f;
            accS[n][j] = e;
            sum += e;
        }
        #pragma unroll
        for (int d = 8; d >= 1; d >>= 1) sum += __shfl_xor(sum, d, 32);
        float inv = 1.0f / sum;
        bf16* pbase = sP + (size_t)t * P_STRIDE + nlo;
        #pragma unroll
        for (int n = 0; n < 16; ++n)
            pbase[n * 16] = (bf16)(accS[n][j] * inv);
    }

    __syncthreads();   // publish sP and sVT to all waves

    // ---- GEMM 3: O = P @ V  (K = 256, N = 64) -------------------------------
    v8f accO[4];
    #pragma unroll
    for (int n = 0; n < 4; ++n) accO[n] = (v8f)0.f;
    const bf16* prow = sP + (size_t)(t0 + nlo) * P_STRIDE + aoff;
    const bf16* vcol = sVT + (size_t)nlo * VT_STRIDE + boff;
    #pragma unroll
    for (int kk = 0; kk < KS_T; ++kk) {
        v16bf a = cat8(*(const v8bf*)(prow + kk * 32),
                       *(const v8bf*)(prow + kk * 32 + 8));
        #pragma unroll
        for (int n = 0; n < 4; ++n) {
            const bf16* colb = vcol + n * 16 * VT_STRIDE + kk * 32;
            v16bf bt = cat8(*(const v8bf*)colb, *(const v8bf*)(colb + 8));
            accO[n] = __builtin_amdgcn_wmma_f32_16x16x32_bf16(
                false, a, false, bt, (short)0, accO[n], false, false);
        }
    }

    // ---- coalesced fp32 store of out[b][t][h] -------------------------------
    float* obase = out + ((size_t)b * Tn + t0 + 8 * half) * HSn + nlo;
    #pragma unroll
    for (int n = 0; n < 4; ++n)
        #pragma unroll
        for (int j = 0; j < 8; ++j)
            obase[j * HSn + n * 16] = accO[n][j];
}

// ---------------------------------------------------------------------------
extern "C" void kernel_launch(void* const* d_in, const int* in_sizes, int n_in,
                              void* d_out, int out_size, void* d_ws, size_t ws_size,
                              hipStream_t stream) {
    const float* x   = (const float*)d_in[0];
    const float* Wq  = (const float*)d_in[1];
    const float* Wk  = (const float*)d_in[2];
    const float* Wv  = (const float*)d_in[3];
    const float* Wql = (const float*)d_in[4];
    const float* Wkl = (const float*)d_in[5];
    float* out = (float*)d_out;

    bf16* Mt  = (bf16*)d_ws;             // [256][384] bf16 = 192 KB (lives in L2)
    bf16* WvT = Mt + Tn * Cn;            // [64][384]  bf16 = 48 KB

    // Precompute collapsed score matrix M^T and W_v^T (tiny: ~13 MFLOP).
    prep_kernel<<<(Tn * Cn + HSn * Cn + 255) / 256, 256, 0, stream>>>(
        Wq, Wk, Wv, Wql, Wkl, Mt, WvT);

    // One workgroup per batch element; 16 waves x 16-row strips; ~165 KB LDS.
    size_t smem = (size_t)SMEM_ELTS * sizeof(bf16);
    head_main<<<Bn, 512, smem, stream>>>(x, Mt, WvT, out);
}
